// VoxelNet_64287070487016
// MI455X (gfx1250) — matrix operand order repaired and checked by reading, not compiled, not run on hardware
//
#include <hip/hip_runtime.h>
#include <hip/hip_bf16.h>

typedef __attribute__((ext_vector_type(16))) _Float16 v16h;
typedef __attribute__((ext_vector_type(8)))  float    v8f;

// 1/sqrt(1 + 1e-3)  (BatchNorm eval scale)
#define BN_RS 0.9995003746f

union AV {
    v16h v;
    float4 q[2];
    _Float16 h[16];
};

// ---- WMMA fragment loaders (f16, 16x16x32) ------------------------------
// A operand (16xK=32): lane L holds row M=L%16.
//   lanes 0-15 : elements 0-7 = K 0..7, elements 8-15 = K 16..23
//   lanes 16-31: elements 0-7 = K 8..15, elements 8-15 = K 24..31
__device__ __forceinline__ void loadA(AV& a, const _Float16* rowBase, bool hi) {
    const _Float16* p = rowBase + (hi ? 8 : 0);
    a.q[0] = *reinterpret_cast<const float4*>(p);
    a.q[1] = *reinterpret_cast<const float4*>(p + 16);
}
// B operand (K=32 x 16): lane L holds column N=L%16, same K order as A
// (lanes 0-15: K 0..7,16..23; lanes 16-31: K 8..15,24..31).
__device__ __forceinline__ void loadB(AV& b, const _Float16* colBase, bool hi) {
    const _Float16* p = colBase + (hi ? 8 : 0);
    b.q[0] = *reinterpret_cast<const float4*>(p);
    b.q[1] = *reinterpret_cast<const float4*>(p + 16);
}

__device__ __forceinline__ void zeroA(AV& a) {
    a.q[0] = make_float4(0.f, 0.f, 0.f, 0.f);
    a.q[1] = make_float4(0.f, 0.f, 0.f, 0.f);
}

// =========================================================================
// Weight packing
// =========================================================================
__global__ void pack_f32_to_f16(const float* __restrict__ src,
                                _Float16* __restrict__ dst, int n) {
    int i = blockIdx.x * blockDim.x + threadIdx.x;
    if (i < n) dst[i] = (_Float16)src[i];
}

// Pack conv weights [O=64, I=CIN, 3,3,3] (f32) into WMMA-B order:
// wp[((tap*NCH + c)*64 + u)*32 + s]  where cin = 32*c + s, tap=(kd*3+kh)*3+kw
template<int CIN>
__global__ void pack_conv_w(const float* __restrict__ w, _Float16* __restrict__ wp) {
    constexpr int NCH = CIN / 32;
    int i = blockIdx.x * blockDim.x + threadIdx.x;
    if (i >= 27 * 64 * CIN) return;
    int s   = i & 31;
    int t   = i >> 5;
    int u   = t % 64;
    int q   = t / 64;
    int c   = q % NCH;
    int tap = q / NCH;
    int cin = c * 32 + s;
    wp[i] = (_Float16)w[((size_t)u * CIN + cin) * 27 + tap];
}

// =========================================================================
// VFE stage 1: mean-subtract, 7->16 matmul, bn+relu, max over T, concat, mask
// One wave per voxel. T = 35, output x1: [N*35, 32] f16.
// =========================================================================
__global__ void vfe1_kernel(const float* __restrict__ feats,
                            const int* __restrict__ nvox,
                            const float* __restrict__ W1,
                            const float* __restrict__ g1,
                            const float* __restrict__ b1,
                            _Float16* __restrict__ x1, int N) {
    int wave = blockIdx.x * (blockDim.x >> 5) + (threadIdx.x >> 5);
    if (wave >= N) return;
    int lane = threadIdx.x & 31;

    const float* fv = feats + (size_t)wave * 35 * 4;
    int nv = nvox[wave];

    // mean over all 35 points of xyz
    float sx = 0.f, sy = 0.f, sz = 0.f;
    for (int t = lane; t < 35; t += 32) {
        sx += fv[t * 4 + 0];
        sy += fv[t * 4 + 1];
        sz += fv[t * 4 + 2];
    }
    for (int m = 16; m >= 1; m >>= 1) {
        sx += __shfl_xor(sx, m, 32);
        sy += __shfl_xor(sy, m, 32);
        sz += __shfl_xor(sz, m, 32);
    }
    float inv = 1.0f / (float)nv;
    float mx = sx * inv, my = sy * inv, mz = sz * inv;

    float h[2][16];
    #pragma unroll
    for (int rep = 0; rep < 2; rep++) {
        int t = lane + 32 * rep;
        float f7[7] = {0.f, 0.f, 0.f, 0.f, 0.f, 0.f, 0.f};
        if (t < 35) {
            f7[0] = fv[t * 4 + 0];
            f7[1] = fv[t * 4 + 1];
            f7[2] = fv[t * 4 + 2];
            f7[3] = fv[t * 4 + 3];
            f7[4] = f7[0] - mx;
            f7[5] = f7[1] - my;
            f7[6] = f7[2] - mz;
        }
        #pragma unroll
        for (int u = 0; u < 16; u++) {
            float acc = 0.f;
            #pragma unroll
            for (int c = 0; c < 7; c++) acc += f7[c] * W1[u * 7 + c];
            acc = acc * (g1[u] * BN_RS) + b1[u];
            h[rep][u] = (t < 35) ? (acc > 0.f ? acc : 0.f) : 0.f;
        }
    }
    // max over all 35 t (values are >= 0, so 0 is a safe identity)
    float cmax[16];
    #pragma unroll
    for (int u = 0; u < 16; u++)
        cmax[u] = fmaxf(h[0][u], (lane < 3) ? h[1][u] : 0.f);
    for (int m = 16; m >= 1; m >>= 1) {
        #pragma unroll
        for (int u = 0; u < 16; u++)
            cmax[u] = fmaxf(cmax[u], __shfl_xor(cmax[u], m, 32));
    }
    #pragma unroll
    for (int rep = 0; rep < 2; rep++) {
        int t = lane + 32 * rep;
        if (t < 35) {
            bool mk = t < nv;
            _Float16* o = x1 + ((size_t)wave * 35 + t) * 32;
            #pragma unroll
            for (int u = 0; u < 16; u++) {
                o[u]      = mk ? (_Float16)h[rep][u] : (_Float16)0.f;
                o[16 + u] = mk ? (_Float16)cmax[u]   : (_Float16)0.f;
            }
        }
    }
}

// =========================================================================
// WMMA GEMM: Y[M,U] = relu(bn(X[M,K] @ Wp[U,K]^T)), f16 in/out, f32 accum.
// Weights staged in LDS once per block (shared by all 8 waves).
// One wave computes a 16-row tile x all U columns. M = 700000.
// =========================================================================
template<int K, int U>
__global__ void gemm_bn_relu(const _Float16* __restrict__ X,
                             const _Float16* __restrict__ Wp,
                             const float* __restrict__ g,
                             const float* __restrict__ bias,
                             _Float16* __restrict__ Y, int numTiles) {
    extern __shared__ char smem[];
    _Float16* lw = (_Float16*)smem;
    // cooperative stage of the full [U,K] f16 weight matrix into LDS
    for (int i = threadIdx.x; i < U * K / 8; i += blockDim.x)
        ((float4*)lw)[i] = ((const float4*)Wp)[i];
    __syncthreads();

    int wave = blockIdx.x * (blockDim.x >> 5) + (threadIdx.x >> 5);
    if (wave >= numTiles) return;  // after barrier: safe
    int lane = threadIdx.x & 31;
    bool hi  = lane >= 16;
    int col  = lane & 15;

    v8f acc[U / 16] = {};
    size_t row = (size_t)wave * 16 + col;
    const _Float16* xrow = X + row * K;

    #pragma unroll
    for (int kc = 0; kc < K; kc += 32) {
        AV a;
        loadA(a, xrow + kc, hi);
        #pragma unroll
        for (int ut = 0; ut < U / 16; ut++) {
            int u = ut * 16 + col;
            AV b;
            loadB(b, lw + (size_t)u * K + kc, hi);
            acc[ut] = __builtin_amdgcn_wmma_f32_16x16x32_f16(
                false, a.v, false, b.v, (short)0, acc[ut], false, false);
        }
    }
    #pragma unroll
    for (int ut = 0; ut < U / 16; ut++) {
        int u = ut * 16 + col;
        float s = g[u] * BN_RS, bb = bias[u];
        #pragma unroll
        for (int r = 0; r < 8; r++) {
            size_t m = (size_t)wave * 16 + r + (hi ? 8 : 0);
            float v = acc[ut][r] * s + bb;
            v = v > 0.f ? v : 0.f;
            Y[m * U + u] = (_Float16)v;
        }
    }
}

// =========================================================================
// VFE concat: x2[n,t,0:64] = h2*mask, x2[n,t,64:128] = maxT(h2)*mask
// =========================================================================
__global__ void vfe_concat(const _Float16* __restrict__ h2,
                           const int* __restrict__ nvox,
                           _Float16* __restrict__ x2, int total) {
    int i = blockIdx.x * blockDim.x + threadIdx.x;
    if (i >= total) return;
    int u = i & 63, n = i >> 6;
    int nv = nvox[n];
    const _Float16* hp = h2 + (size_t)n * 35 * 64 + u;
    float mx = 0.f;
    for (int t = 0; t < 35; t++) {
        float v = (float)hp[t * 64];
        mx = v > mx ? v : mx;
    }
    _Float16* xp = x2 + (size_t)n * 35 * 128 + u;
    _Float16 mh = (_Float16)mx, zh = (_Float16)0.f;
    for (int t = 0; t < 35; t++) {
        bool mk = t < nv;
        xp[t * 128]      = mk ? hp[t * 64] : zh;
        xp[t * 128 + 64] = mk ? mh : zh;
    }
}

// =========================================================================
// Per-voxel max over valid T + scatter into dense grid [10,400,352,128] f16
// =========================================================================
__global__ void max_scatter(const _Float16* __restrict__ y,
                            const int* __restrict__ nvox,
                            const int* __restrict__ coors,
                            _Float16* __restrict__ grid, int total) {
    int i = blockIdx.x * blockDim.x + threadIdx.x;
    if (i >= total) return;
    int c = i & 127, n = i >> 7;
    int nv = nvox[n];
    const _Float16* yp = y + (size_t)n * 35 * 128 + c;
    float mx = 0.f;
    for (int t = 0; t < nv; t++) {
        float v = (float)yp[t * 128];
        mx = v > mx ? v : mx;
    }
    int d  = coors[n * 4 + 1];
    int hh = coors[n * 4 + 2];
    int ww = coors[n * 4 + 3];
    grid[(((size_t)d * 400 + hh) * 352 + ww) * 128 + c] = (_Float16)mx;
}

// =========================================================================
// Implicit-GEMM 3x3x3 conv + bn + relu, WMMA f16->f32.
// Input DHWC f16. One wave: 16 consecutive W positions x all COUT channels.
// Weights for one kd-slice (9 taps) staged in LDS per block; 16 waves/block
// share them (wave counts are exact multiples of 16 -> no guards, barriers
// are uniform, EXEC stays all-ones for WMMA).
// Stride (SD,1,1), pad (PD,1,1). If finalMode: write f32 (c*Do+d, h, w).
// =========================================================================
template<int CIN, int COUT, int SD, int PD>
__global__ void conv3d_bn_relu(const _Float16* __restrict__ in, int Din,
                               const _Float16* __restrict__ wp,
                               const float* __restrict__ g,
                               const float* __restrict__ bias,
                               _Float16* __restrict__ outH,
                               float* __restrict__ outF,
                               int Do, int finalMode) {
    constexpr int NCH   = CIN / 32;
    constexpr int SLICE = 9 * NCH * 64 * 32;  // halves per kd slice
    extern __shared__ char smem[];
    _Float16* lw = (_Float16*)smem;

    int wave = blockIdx.x * (blockDim.x >> 5) + (threadIdx.x >> 5);
    int lane = threadIdx.x & 31;
    bool hi  = lane >= 16;
    int col  = lane & 15;

    int wt  = wave % 22;
    int tmp = wave / 22;
    int ho  = tmp % 400;
    int dz  = tmp / 400;
    int w0  = wt * 16;
    int aw  = w0 + col;  // this lane's A-row (output w position)

    v8f acc[COUT / 16] = {};

    for (int kd = 0; kd < 3; kd++) {
        // stage this kd's 9-tap weight slice into LDS (all threads)
        __syncthreads();
        const float4* gsrc = (const float4*)(wp + (size_t)kd * SLICE);
        for (int i = threadIdx.x; i < SLICE / 8; i += blockDim.x)
            ((float4*)lw)[i] = gsrc[i];
        __syncthreads();

        int di = dz * SD + kd - PD;
        bool okd = (di >= 0) && (di < Din);
        for (int kh = 0; kh < 3; kh++) {
            int hh = ho + kh - 1;
            bool okh = okd && (hh >= 0) && (hh < 400);
            #pragma unroll
            for (int kw = 0; kw < 3; kw++) {
                int wi = aw + kw - 1;
                bool ok = okh && (wi >= 0) && (wi < 352);
                int tin = kh * 3 + kw;  // tap index within slice
                size_t vbase = (((size_t)di * 400 + hh) * 352 + wi) * CIN;
                #pragma unroll
                for (int c = 0; c < NCH; c++) {
                    AV a;
                    if (ok) loadA(a, in + vbase + c * 32, hi);
                    else    zeroA(a);
                    const _Float16* wc = lw + (((size_t)tin * NCH + c) * 64) * 32;
                    #pragma unroll
                    for (int ut = 0; ut < COUT / 16; ut++) {
                        int u = ut * 16 + col;
                        AV b;
                        loadB(b, wc + u * 32, hi);
                        acc[ut] = __builtin_amdgcn_wmma_f32_16x16x32_f16(
                            false, a.v, false, b.v, (short)0, acc[ut], false, false);
                    }
                }
            }
        }
    }

    #pragma unroll
    for (int ut = 0; ut < COUT / 16; ut++) {
        int u = ut * 16 + col;
        float s = g[u] * BN_RS, bb = bias[u];
        #pragma unroll
        for (int r = 0; r < 8; r++) {
            int wo = w0 + r + (hi ? 8 : 0);
            float v = acc[ut][r] * s + bb;
            v = v > 0.f ? v : 0.f;
            if (finalMode)
                outF[(((size_t)u * Do + dz) * 400 + ho) * 352 + wo] = v;
            else
                outH[((((size_t)dz * 400 + ho) * 352 + wo) * COUT) + u] = (_Float16)v;
        }
    }
}

// =========================================================================
extern "C" void kernel_launch(void* const* d_in, const int* in_sizes, int n_in,
                              void* d_out, int out_size, void* d_ws, size_t ws_size,
                              hipStream_t stream) {
    (void)in_sizes; (void)n_in; (void)out_size; (void)ws_size;

    const float* feats = (const float*)d_in[0];
    const int*   nvox  = (const int*)d_in[1];
    const int*   coors = (const int*)d_in[2];
    const float* w1  = (const float*)d_in[4];
    const float* g1  = (const float*)d_in[5];
    const float* b1  = (const float*)d_in[6];
    const float* w2  = (const float*)d_in[7];
    const float* g2  = (const float*)d_in[8];
    const float* b2  = (const float*)d_in[9];
    const float* wl  = (const float*)d_in[10];
    const float* gl  = (const float*)d_in[11];
    const float* bl  = (const float*)d_in[12];
    const float* wc1 = (const float*)d_in[13];
    const float* gc1 = (const float*)d_in[14];
    const float* bc1 = (const float*)d_in[15];
    const float* wc2 = (const float*)d_in[16];
    const float* gc2 = (const float*)d_in[17];
    const float* bc2 = (const float*)d_in[18];
    const float* wc3 = (const float*)d_in[19];
    const float* gc3 = (const float*)d_in[20];
    const float* bc3 = (const float*)d_in[21];
    float* out = (float*)d_out;

    // ---- workspace layout (lifetime-aliased) ----
    constexpr size_t MB = 1ull << 20;
    char* ws = (char*)d_ws;
    _Float16* x1   = (_Float16*)(ws + 0);          // [700000,32]  44.8 MB
    _Float16* h2   = (_Float16*)(ws + 48 * MB);    // [700000,64]  89.6 MB
    _Float16* x2   = (_Float16*)(ws + 184 * MB);   // [700000,128] 179.2 MB
    _Float16* y    = (_Float16*)(ws + 0);          // [700000,128] (aliases x1+h2)
    _Float16* r1   = (_Float16*)(ws + 0);          // [5,400,352,64] (aliases y)
    _Float16* r2   = (_Float16*)(ws + 96 * MB);    // [3,400,352,64]
    _Float16* grid = (_Float16*)(ws + 368 * MB);   // [10,400,352,128] 360.4 MB
    _Float16* wp2  = (_Float16*)(ws + 732 * MB);   // vfe2 weights f16
    _Float16* wpl  = (_Float16*)(ws + 733 * MB);   // lin  weights f16
    _Float16* wpc1 = (_Float16*)(ws + 734 * MB);   // conv1 packed
    _Float16* wpc2 = (_Float16*)(ws + 735 * MB);   // conv2 packed
    _Float16* wpc3 = (_Float16*)(ws + 736 * MB);   // conv3 packed

    const int B = 256;
    auto cdiv = [](int a, int b) { return (a + b - 1) / b; };

    // 1) weight packing
    pack_f32_to_f16<<<cdiv(64 * 32, B), B, 0, stream>>>(w2, wp2, 64 * 32);
    pack_f32_to_f16<<<cdiv(128 * 128, B), B, 0, stream>>>(wl, wpl, 128 * 128);
    pack_conv_w<128><<<cdiv(27 * 64 * 128, B), B, 0, stream>>>(wc1, wpc1);
    pack_conv_w<64><<<cdiv(27 * 64 * 64, B), B, 0, stream>>>(wc2, wpc2);
    pack_conv_w<64><<<cdiv(27 * 64 * 64, B), B, 0, stream>>>(wc3, wpc3);

    // 2) zero the dense grid (scatter target)
    hipMemsetAsync(grid, 0, (size_t)10 * 400 * 352 * 128 * sizeof(_Float16), stream);

    // 3) VFE layer 1 (one wave per voxel)
    vfe1_kernel<<<cdiv(20000, 8), B, 0, stream>>>(feats, nvox, w1, g1, b1, x1, 20000);

    // 4) VFE layer 2 GEMM: [700000,32] @ [64,32]^T ; LDS-staged weights (4 KB)
    const int tiles = 700000 / 16;
    gemm_bn_relu<32, 64><<<cdiv(tiles, 8), B, 64 * 32 * 2, stream>>>(
        x1, wp2, g2, b2, h2, tiles);

    // 5) max/concat/mask -> x2 [700000,128]
    vfe_concat<<<cdiv(20000 * 64, B), B, 0, stream>>>(h2, nvox, x2, 20000 * 64);

    // 6) linear GEMM: [700000,128] @ [128,128]^T ; LDS-staged weights (32 KB)
    gemm_bn_relu<128, 128><<<cdiv(tiles, 8), B, 128 * 128 * 2, stream>>>(
        x2, wpl, gl, bl, y, tiles);

    // 7) per-voxel max over valid T + scatter to grid
    max_scatter<<<cdiv(20000 * 128, B), B, 0, stream>>>(y, nvox, coors, grid, 20000 * 128);

    // conv kernels: 512 threads = 16 waves/block, wave counts divide exactly.
    // LDS slice bytes: 9 taps * NCH * 64 * 32 halves * 2B
    // 8) conv1: [10,400,352,128] -> [5,400,352,64], stride(2,1,1), pad 1
    conv3d_bn_relu<128, 64, 2, 1><<<44000 / 16, 512, 9 * 4 * 64 * 32 * 2, stream>>>(
        grid, 10, wpc1, gc1, bc1, r1, nullptr, 5, 0);

    // 9) conv2: [5,...,64] -> [3,400,352,64], stride 1, pad (0,1,1)
    conv3d_bn_relu<64, 64, 1, 0><<<26400 / 16, 512, 9 * 2 * 64 * 32 * 2, stream>>>(
        r1, 5, wpc2, gc2, bc2, r2, nullptr, 3, 0);

    // 10) conv3: [3,...,64] -> [2,400,352,64], stride(2,1,1), pad 1; final f32
    conv3d_bn_relu<64, 64, 2, 1><<<17600 / 16, 512, 9 * 2 * 64 * 32 * 2, stream>>>(
        r2, 3, wpc3, gc3, bc3, nullptr, out, 2, 1);
}
